// MinigridPPOLSTMAgent_730144440964
// MI455X (gfx1250) — compile-verified, matrix operand order checked by reading, not run on hardware
//
#include <hip/hip_runtime.h>
#include <hip/hip_bf16.h>
#include <math.h>

// ---------------- problem constants ----------------
#define TT   128
#define BB   256
#define NN   (TT*BB)          // 32768
#define HID  128
#define G4   512              // 4*HID
#define NFLAT 1024

typedef __attribute__((ext_vector_type(16))) _Float16 v16h;
typedef __attribute__((ext_vector_type(8)))  _Float16 v8h;
typedef __attribute__((ext_vector_type(8)))  float    v8f;

// K index inside a 32-wide chunk for 16-bit A/B WMMA fragments (ISA 7.12.2)
__device__ __forceinline__ int frag_k(int e, int g2) {
    return e + 8 * g2 + ((e & 8) ? 8 : 0);
}

__device__ __forceinline__ float sigm(float x) { return 1.0f / (1.0f + __expf(-x)); }

#define SHUF16(lo, hi) __builtin_shufflevector(lo, hi, 0,1,2,3,4,5,6,7,8,9,10,11,12,13,14,15)

// sched_group_barrier masks
#define SGB_WMMA     0x008
#define SGB_VMEMRD   0x020
#define SGB_DSRD     0x100

// ---------------------------------------------------------------------------
// Kernel 1: conv stack, one image per wave, LDS staging, writes feat as f16
// ---------------------------------------------------------------------------
__global__ __launch_bounds__(256)
void conv_kernel(const float* __restrict__ obs,
                 const float* __restrict__ w1, const float* __restrict__ b1,
                 const float* __restrict__ w2, const float* __restrict__ b2,
                 const float* __restrict__ w3, const float* __restrict__ b3,
                 _Float16* __restrict__ feat) {
    __shared__ float s1[8][16 * 6 * 6];   // 18 KB
    __shared__ float s2[8][32 * 5 * 5];   // 25.6 KB
    const int wid  = threadIdx.x >> 5;
    const int lane = threadIdx.x & 31;
    const int n    = blockIdx.x * 8 + wid;
    const float* ob = obs + (size_t)n * 147;   // [7][7][3] NHWC

    // conv1: 3 -> 16, 7x7 -> 6x6
    for (int idx = lane; idx < 576; idx += 32) {
        int o = idx / 36, rem = idx % 36, y = rem / 6, x = rem % 6;
        float acc = b1[o];
        #pragma unroll
        for (int ky = 0; ky < 2; ++ky)
            #pragma unroll
            for (int kx = 0; kx < 2; ++kx)
                #pragma unroll
                for (int c = 0; c < 3; ++c)
                    acc += ob[(y + ky) * 21 + (x + kx) * 3 + c] *
                           w1[((o * 3 + c) * 2 + ky) * 2 + kx];
        s1[wid][idx] = fmaxf(acc, 0.0f);
    }
    __syncthreads();
    // conv2: 16 -> 32, 6x6 -> 5x5
    for (int idx = lane; idx < 800; idx += 32) {
        int o = idx / 25, rem = idx % 25, y = rem / 5, x = rem % 5;
        float acc = b2[o];
        for (int c = 0; c < 16; ++c)
            #pragma unroll
            for (int ky = 0; ky < 2; ++ky)
                #pragma unroll
                for (int kx = 0; kx < 2; ++kx)
                    acc += s1[wid][c * 36 + (y + ky) * 6 + (x + kx)] *
                           w2[((o * 16 + c) * 2 + ky) * 2 + kx];
        s2[wid][idx] = fmaxf(acc, 0.0f);
    }
    __syncthreads();
    // conv3: 32 -> 64, 5x5 -> 4x4, flatten order o*16+y*4+x == idx
    for (int idx = lane; idx < 1024; idx += 32) {
        int o = idx / 16, rem = idx % 16, y = rem / 4, x = rem % 4;
        float acc = b3[o];
        for (int c = 0; c < 32; ++c)
            #pragma unroll
            for (int ky = 0; ky < 2; ++ky)
                #pragma unroll
                for (int kx = 0; kx < 2; ++kx)
                    acc += s2[wid][c * 25 + (y + ky) * 5 + (x + kx)] *
                           w3[((o * 32 + c) * 2 + ky) * 2 + kx];
        feat[(size_t)n * NFLAT + idx] = (_Float16)fmaxf(acc, 0.0f);
    }
}

// ---------------------------------------------------------------------------
// Kernel 2: pack LSTM weights to f16, fragment-ready layout:
//   wP[ tn ][ kc ][ lane ][ e ]  == B-frag element e of lane for tile (tn,kc)
// so each lane's fragment is 32 contiguous bytes (2x global_load_b128).
// ---------------------------------------------------------------------------
__global__ __launch_bounds__(256)
void pack_weights(const float* __restrict__ w_ih, const float* __restrict__ w_hh,
                  _Float16* __restrict__ w_ihP, _Float16* __restrict__ w_hhP) {
    const int p = blockIdx.x * 256 + threadIdx.x;
    if (p < 32 * 32 * 32 * 16) {                 // tn(32) kc(32) lane(32) e(16)
        const int e = p & 15, lane = (p >> 4) & 31, kc = (p >> 9) & 31, tn = p >> 14;
        const int g2 = lane >> 4, lr = lane & 15;
        const int K  = kc * 32 + frag_k(e, g2);
        const int nc = tn * 16 + lr;
        w_ihP[p] = (_Float16)w_ih[(size_t)nc * NFLAT + K];
    }
    if (p < 32 * 4 * 32 * 16) {                  // tn(32) kc(4) lane(32) e(16)
        const int e = p & 15, lane = (p >> 4) & 31, kc = (p >> 9) & 3, tn = p >> 11;
        const int g2 = lane >> 4, lr = lane & 15;
        const int K  = kc * 32 + frag_k(e, g2);
        const int nc = tn * 16 + lr;
        w_hhP[p] = (_Float16)w_hh[nc * HID + K];
    }
}

// ---------------------------------------------------------------------------
// Kernel 3: gates_x = feat @ w_ih^T + (b_ih + b_hh)   [N,512] via WMMA f16
// each wave: one 16-row M tile x 8 N-tiles (A fragment reused 8x).
// sched_group_barrier pins the region to: 18 VMEM reads, then 8 WMMAs, so
// loadcnt waits are staggered instead of full drains before each WMMA.
// ---------------------------------------------------------------------------
__global__ __launch_bounds__(256)
void gemm_gates_x(const _Float16* __restrict__ feat, const _Float16* __restrict__ wP,
                  const float* __restrict__ b_ih, const float* __restrict__ b_hh,
                  float* __restrict__ gx) {
    const int wid  = threadIdx.x >> 5;
    const int lane = threadIdx.x & 31;
    const int wg   = blockIdx.x * 8 + wid;        // 8192 waves
    const int tm   = wg >> 2;                     // 2048 M tiles
    const int ng   = wg & 3;                      // 4 groups of 8 N tiles
    const int g2 = lane >> 4, lr = lane & 15;
    const _Float16* arow = feat + (size_t)(tm * 16 + lr) * NFLAT + 8 * g2;
    const v16h* bbase = (const v16h*)wP + (size_t)(ng * 8 * 32) * 32 + lane;

    v8f acc[8];
    #pragma unroll
    for (int j = 0; j < 8; ++j) acc[j] = (v8f){};

    for (int kc = 0; kc < 32; ++kc) {
        const v16h a = SHUF16(*(const v8h*)(arow + kc * 32),
                              *(const v8h*)(arow + kc * 32 + 16));
        v16h bf[8];
        #pragma unroll
        for (int j = 0; j < 8; ++j)
            bf[j] = bbase[(j * 32 + kc) * 32];
        #pragma unroll
        for (int j = 0; j < 8; ++j)
            acc[j] = __builtin_amdgcn_wmma_f32_16x16x32_f16(
                false, a, false, bf[j], (short)0, acc[j], false, false);
        // scheduling pipeline for this iteration: loads first, then WMMAs
        __builtin_amdgcn_sched_group_barrier(SGB_VMEMRD, 18, 0);
        __builtin_amdgcn_sched_group_barrier(SGB_WMMA,    8, 0);
    }
    #pragma unroll
    for (int j = 0; j < 8; ++j) {
        const int ncol = (ng * 8 + j) * 16 + lr;
        const float bias = b_ih[ncol] + b_hh[ncol];
        #pragma unroll
        for (int r = 0; r < 8; ++r)
            gx[(size_t)(tm * 16 + r + 8 * g2) * G4 + ncol] = acc[j][r] + bias;
    }
}

// ---------------------------------------------------------------------------
// Kernel 4: persistent recurrent LSTM. 1 block x 1024 threads (32 waves).
// h lives in 64KB LDS (f16); c lives in VGPRs. Each wave owns 4 tile-groups
// (16 batch x 16 hid); per group the 4 gate tiles share the C/D layout so
// i/f/g/o for one (b,hid) element sit in the same lane/VGPR.
// (Kept at moderate VGPR use: all 32 waves must fit on one WGP.)
// ---------------------------------------------------------------------------
__global__ __launch_bounds__(1024)
void lstm_kernel(const float* __restrict__ done,
                 const float* __restrict__ h0, const float* __restrict__ c0,
                 const _Float16* __restrict__ whhP,
                 const float* __restrict__ gx,
                 float* __restrict__ hid,
                 float* __restrict__ out) {
    __shared__ __align__(32) _Float16 hbuf[BB * HID];   // 64 KB
    const int tid  = threadIdx.x;
    const int wid  = tid >> 5;
    const int lane = tid & 31;
    const int g2 = lane >> 4, lr = lane & 15;
    const v16h* bbase = (const v16h*)whhP + lane;  // + ((gate*8+tn0)*4+kc)*32

    for (int i = tid; i < BB * HID; i += 1024) hbuf[i] = (_Float16)h0[i];

    int m0s[4], n0s[4];
    float c_reg[4][8], hnew[4][8];
    #pragma unroll
    for (int i = 0; i < 4; ++i) {
        const int gidx = wid * 4 + i;             // 128 tile-groups
        m0s[i] = (gidx >> 3) * 16;                // batch tile
        n0s[i] = (gidx & 7) * 16;                 // hid tile
        #pragma unroll
        for (int r = 0; r < 8; ++r)
            c_reg[i][r] = c0[(m0s[i] + r + 8 * g2) * HID + n0s[i] + lr];
    }
    __syncthreads();

    for (int t = 0; t < TT; ++t) {
        const float* dt  = done + t * BB;
        const float* gxt = gx + (size_t)t * BB * G4;
        #pragma unroll
        for (int i = 0; i < 4; ++i) {
            const int m0 = m0s[i], n0 = n0s[i], tn0 = n0 >> 4;
            v8f acc[4];
            #pragma unroll
            for (int gate = 0; gate < 4; ++gate)
                #pragma unroll
                for (int r = 0; r < 8; ++r)
                    acc[gate][r] = gxt[(size_t)(m0 + r + 8 * g2) * G4 + gate * HID + n0 + lr];

            // mask depends only on the A row (= lane&15) -> scalar per lane
            const _Float16 amh = (_Float16)(1.0f - dt[m0 + lr]);
            const _Float16* hrow = hbuf + (m0 + lr) * HID;

            #pragma unroll
            for (int kc = 0; kc < 4; ++kc) {
                const v8h lo = *(const v8h*)(hrow + kc * 32 + 8 * g2);
                const v8h hi = *(const v8h*)(hrow + kc * 32 + 16 + 8 * g2);
                v16h a = SHUF16(lo, hi);
                #pragma unroll
                for (int e = 0; e < 16; ++e) a[e] = a[e] * amh;
                v16h bf[4];
                #pragma unroll
                for (int gate = 0; gate < 4; ++gate)
                    bf[gate] = bbase[((gate * 8 + tn0) * 4 + kc) * 32];
                #pragma unroll
                for (int gate = 0; gate < 4; ++gate)
                    acc[gate] = __builtin_amdgcn_wmma_f32_16x16x32_f16(
                        false, a, false, bf[gate], (short)0, acc[gate], false, false);
                // per K-chunk pipeline: DS reads, VMEM reads, then WMMAs
                __builtin_amdgcn_sched_group_barrier(SGB_DSRD,   2, 0);
                __builtin_amdgcn_sched_group_barrier(SGB_VMEMRD, 8, 0);
                __builtin_amdgcn_sched_group_barrier(SGB_WMMA,   4, 0);
            }
            #pragma unroll
            for (int r = 0; r < 8; ++r) {
                const int bi = m0 + r + 8 * g2;
                const float m  = 1.0f - dt[bi];
                const float ig = sigm(acc[0][r]);
                const float fg = sigm(acc[1][r]);
                const float gg = tanhf(acc[2][r]);
                const float og = sigm(acc[3][r]);
                const float c  = fg * (m * c_reg[i][r]) + ig * gg;
                c_reg[i][r] = c;
                const float h = og * tanhf(c);
                hnew[i][r] = h;
                hid[((size_t)t * BB + bi) * HID + n0 + lr] = h;
            }
        }
        __syncthreads();   // all reads of hbuf for step t complete
        #pragma unroll
        for (int i = 0; i < 4; ++i)
            #pragma unroll
            for (int r = 0; r < 8; ++r)
                hbuf[(m0s[i] + r + 8 * g2) * HID + n0s[i] + lr] = (_Float16)hnew[i][r];
        __syncthreads();   // new h visible before step t+1
    }

    // hT, cT tails of d_out
    float* hT = out + (size_t)NN * 8;
    float* cT = hT + BB * HID;
    #pragma unroll
    for (int i = 0; i < 4; ++i)
        #pragma unroll
        for (int r = 0; r < 8; ++r) {
            const int bi = m0s[i] + r + 8 * g2;
            hT[bi * HID + n0s[i] + lr] = hnew[i][r];
            cT[bi * HID + n0s[i] + lr] = c_reg[i][r];
        }
}

// ---------------------------------------------------------------------------
// Kernel 5: actor/critic heads, one thread per output element (128 MACs)
// ---------------------------------------------------------------------------
__global__ __launch_bounds__(256)
void heads_kernel(const float* __restrict__ hid,
                  const float* __restrict__ aw, const float* __restrict__ ab,
                  const float* __restrict__ cw, const float* __restrict__ cb,
                  float* __restrict__ out) {
    const int gid = blockIdx.x * 256 + threadIdx.x;   // [0, N*8)
    const int n = gid >> 3, col = gid & 7;
    const float* h = hid + (size_t)n * HID;
    const float* w;
    float acc;
    if (col < 7) { w = aw + col * HID; acc = ab[col]; }
    else         { w = cw;             acc = cb[0];   }
    #pragma unroll 8
    for (int k = 0; k < HID; ++k) acc += h[k] * w[k];
    out[gid] = acc;
}

// ---------------------------------------------------------------------------
extern "C" void kernel_launch(void* const* d_in, const int* in_sizes, int n_in,
                              void* d_out, int out_size, void* d_ws, size_t ws_size,
                              hipStream_t stream) {
    (void)in_sizes; (void)n_in; (void)out_size; (void)ws_size;
    const float* obs      = (const float*)d_in[0];
    const float* done     = (const float*)d_in[1];
    const float* h0       = (const float*)d_in[2];
    const float* c0       = (const float*)d_in[3];
    const float* conv1_w  = (const float*)d_in[4];
    const float* conv1_b  = (const float*)d_in[5];
    const float* conv2_w  = (const float*)d_in[6];
    const float* conv2_b  = (const float*)d_in[7];
    const float* conv3_w  = (const float*)d_in[8];
    const float* conv3_b  = (const float*)d_in[9];
    const float* w_ih     = (const float*)d_in[10];
    const float* w_hh     = (const float*)d_in[11];
    const float* b_ih     = (const float*)d_in[12];
    const float* b_hh     = (const float*)d_in[13];
    const float* actor_w  = (const float*)d_in[14];
    const float* actor_b  = (const float*)d_in[15];
    const float* critic_w = (const float*)d_in[16];
    const float* critic_b = (const float*)d_in[17];
    float* out = (float*)d_out;

    // workspace layout
    char* ws = (char*)d_ws;
    _Float16* feat  = (_Float16*)ws;                                   // 64 MB
    _Float16* w_ihP = (_Float16*)(ws + (size_t)NN * NFLAT * 2);        // 1 MB
    _Float16* w_hhP = (_Float16*)((char*)w_ihP + (size_t)NFLAT * G4 * 2);
    float*    gx    = (float*)((char*)w_hhP + (size_t)HID * G4 * 2);   // 64 MB
    float*    hid   = (float*)((char*)gx + (size_t)NN * G4 * 4);       // 16 MB

    // 1) conv features (8 images per 256-thread block)
    conv_kernel<<<NN / 8, 256, 0, stream>>>(obs, conv1_w, conv1_b, conv2_w, conv2_b,
                                            conv3_w, conv3_b, feat);
    // 2) pack weights into fragment-ready layout
    pack_weights<<<(NFLAT * G4 + 255) / 256, 256, 0, stream>>>(w_ih, w_hh, w_ihP, w_hhP);
    // 3) big X-projection GEMM (+bias): 8192 waves, 8 N-tiles per wave
    gemm_gates_x<<<8192 / 8, 256, 0, stream>>>(feat, w_ihP, b_ih, b_hh, gx);
    // 4) persistent recurrent LSTM on one WGP
    lstm_kernel<<<1, 1024, 0, stream>>>(done, h0, c0, w_hhP, gx, hid, out);
    // 5) heads
    heads_kernel<<<(NN * 8) / 256, 256, 0, stream>>>(hid, actor_w, actor_b,
                                                     critic_w, critic_b, out);
}
